// RNN_Embedding_16020228014874
// MI455X (gfx1250) — matrix operand order repaired and checked by reading, not compiled, not run on hardware
//
#include <hip/hip_runtime.h>
#include <hip/hip_bf16.h>

typedef __bf16 bf16;
typedef __attribute__((ext_vector_type(16))) __bf16 bf16x16;
typedef __attribute__((ext_vector_type(8)))  __bf16 bf16x8;
typedef __attribute__((ext_vector_type(4)))  __bf16 bf16x4;
typedef __attribute__((ext_vector_type(8)))  float  f32x8;

#define B_    128
#define IN_   512
#define HID_  512
#define OUT_  512
#define T_    512
#define G3_   1536

#define NWG   64
#define NTHR  256
#define NWAVE ((NWG * NTHR) / 32)   // 512 waves

// ---------------------------------------------------------------------------
// Grid-wide sense-reversing barrier (agent-scope atomics).
// ---------------------------------------------------------------------------
__device__ __forceinline__ void grid_sync(unsigned* bar, unsigned idx) {
  __syncthreads();
  if (threadIdx.x == 0) {
    __threadfence();
    unsigned* cnt = &bar[idx & 1u];
    unsigned* gen = &bar[2];
    unsigned g = __hip_atomic_load(gen, __ATOMIC_ACQUIRE, __HIP_MEMORY_SCOPE_AGENT);
    unsigned prev = __hip_atomic_fetch_add(cnt, 1u, __ATOMIC_ACQ_REL, __HIP_MEMORY_SCOPE_AGENT);
    if (prev == (unsigned)NWG - 1u) {
      __hip_atomic_store(cnt, 0u, __ATOMIC_RELAXED, __HIP_MEMORY_SCOPE_AGENT);
      __hip_atomic_fetch_add(gen, 1u, __ATOMIC_RELEASE, __HIP_MEMORY_SCOPE_AGENT);
    } else {
      while (__hip_atomic_load(gen, __ATOMIC_ACQUIRE, __HIP_MEMORY_SCOPE_AGENT) == g) {
        __builtin_amdgcn_s_sleep(1);
      }
    }
    __threadfence();
  }
  __syncthreads();
}

// ---------------------------------------------------------------------------
// Async staging: 16KB contiguous strip (16 rows x 512 bf16) global -> LDS.
// 256 threads x 4 passes x 16B via GLOBAL_LOAD_ASYNC_TO_LDS_B128 (ASYNCcnt).
// ---------------------------------------------------------------------------
__device__ __forceinline__ void stage_strip_async(bf16* sm, const bf16* g) {
  unsigned lbase = (unsigned)(size_t)sm + threadIdx.x * 16u;
  const char* gp = (const char*)g + threadIdx.x * 16u;
  #pragma unroll
  for (int p = 0; p < 4; ++p) {
    asm volatile("global_load_async_to_lds_b128 %0, %1, off"
                 :: "v"(lbase + (unsigned)(p * 4096)), "v"(gp + p * 4096)
                 : "memory");
  }
}

__device__ __forceinline__ void wait_async_lds() {
  asm volatile("s_wait_asynccnt 0x0" ::: "memory");
}

// ---------------------------------------------------------------------------
// WMMA bf16 16x16x32 operand loaders (contiguous k order of this code, same for A and B).
//  A: lane L -> row M = L&15; K elems (L&16?16:0)+0..15 (contiguous)
//  B: lane L -> col N = L&15; K elems (L&16?16:0)+0..15 (contiguous)
//  C/D: lane L -> col N = L&15; VGPR r -> row M = r + ((L&16)?8:0)
// ---------------------------------------------------------------------------
__device__ __forceinline__ bf16x16 load_a_lds(const bf16* S, int k0, int lane) {
  int m  = lane & 15;
  int kb = k0 + ((lane & 16) ? 16 : 0);
  const bf16* p = S + m * 512 + kb;          // strip row stride fixed at 512
  bf16x8 lo = *(const bf16x8*)(p);
  bf16x8 hi = *(const bf16x8*)(p + 8);
  return __builtin_shufflevector(lo, hi, 0,1,2,3,4,5,6,7,8,9,10,11,12,13,14,15);
}

__device__ __forceinline__ bf16x16 load_b32x16(const bf16* W, int ldw, int n0, int k0, int lane) {
  int n  = n0 + (lane & 15);
  int kb = k0 + ((lane & 16) ? 16 : 0);
  return *(const bf16x16*)(W + (size_t)n * ldw + kb);
}

// acc += S_lds[0:16, 0:K] * W[n0:+16, 0:K]^T   (A from LDS strip, W from L2)
__device__ __forceinline__ f32x8 wmma_acc_lds(f32x8 acc, const bf16* S,
                                              const bf16* __restrict__ W, int ldw,
                                              int n0, int K, int lane) {
  #pragma unroll
  for (int k0 = 0; k0 < K; k0 += 32) {
    bf16x16 a = load_a_lds(S, k0, lane);
    bf16x16 b = load_b32x16(W, ldw, n0, k0, lane);
    acc = __builtin_amdgcn_wmma_f32_16x16x32_bf16(false, a, false, b, (short)0, acc, false, false);
  }
  return acc;
}

__device__ __forceinline__ void store_tile_f32(float* C, int ldc, int m0, int n0, f32x8 acc, int lane) {
  int n  = n0 + (lane & 15);
  int mb = m0 + ((lane & 16) ? 8 : 0);
  #pragma unroll
  for (int r = 0; r < 8; ++r) C[(size_t)(mb + r) * ldc + n] = acc[r];
}

// ---------------------------------------------------------------------------
// Prep kernels
// ---------------------------------------------------------------------------
__global__ void cvt_bf16(const float* __restrict__ s, bf16* __restrict__ d, int n) {
  int i = blockIdx.x * blockDim.x + threadIdx.x;
  if (i < n) d[i] = (bf16)s[i];
}

__global__ __launch_bounds__(256) void xprep(const float* __restrict__ x,
                                             bf16* __restrict__ xbf,
                                             bf16* __restrict__ xT) {
  __shared__ float tile[32][33];
  int b  = blockIdx.x;
  int it = blockIdx.y;
  int tt = blockIdx.z;
  int tx = threadIdx.x & 31;
  int ty = threadIdx.x >> 5;
  const size_t xbase = (size_t)b * IN_ * T_;
  #pragma unroll
  for (int r = 0; r < 4; ++r) {
    int i    = it * 32 + ty + r * 8;
    int tcol = tt * 32 + tx;
    float v = x[xbase + (size_t)i * T_ + tcol];
    tile[ty + r * 8][tx] = v;
    xbf[xbase + (size_t)i * T_ + tcol] = (bf16)v;
  }
  __syncthreads();
  #pragma unroll
  for (int r = 0; r < 4; ++r) {
    int trow = tt * 32 + ty + r * 8;
    int icol = it * 32 + tx;
    xT[((size_t)trow * B_ + b) * IN_ + icol] = (bf16)tile[tx][ty + r * 8];
  }
}

__global__ void init_state(float* h, bf16* hbf, unsigned* bar, int n) {
  int i = blockIdx.x * blockDim.x + threadIdx.x;
  if (i < n) { h[i] = 0.0f; hbf[i] = (bf16)0.0f; }
  if (i < 8) bar[i] = 0u;
}

// ---------------------------------------------------------------------------
// Persistent sequential-scan kernel.
// GEMM phases: each WG owns one 16-row A strip (LDS-staged, async) and 4
// output tiles; 2 waves per tile split K into its two natural halves and
// combine partials through LDS.
// ---------------------------------------------------------------------------
__global__ __launch_bounds__(NTHR) void rnn_persistent(
    const bf16* __restrict__ xbf,    // [B, IN, T]
    const bf16* __restrict__ xT,     // [T, B, IN]
    const bf16* __restrict__ attnW,  // [512, 1024]
    const float* __restrict__ attnB,
    const bf16* __restrict__ combW,  // [512, 1024]
    const float* __restrict__ combB,
    const bf16* __restrict__ wih,    // [1536, 512]
    const float* __restrict__ bih,
    const bf16* __restrict__ whh,    // [1536, 512]
    const float* __restrict__ bhh,
    const bf16* __restrict__ outW,   // [512, 512]
    const float* __restrict__ outB,
    float* __restrict__ h,           // [B, 512]
    bf16*  __restrict__ hbf,         // [B, 512]
    float* __restrict__ attn,        // [B, 512] logits -> softmax in place
    bf16*  __restrict__ appliedbf,   // [B, 512]
    bf16*  __restrict__ gbf,         // [B, 512]
    float* __restrict__ gi,          // [B, 1536]
    float* __restrict__ ghm,         // [B, 1536]
    float* __restrict__ out,         // [B, 512]
    unsigned* __restrict__ bar) {

  __shared__ bf16  smA0[16 * 512];   // 16KB strip
  __shared__ bf16  smA1[16 * 512];   // 16KB strip
  __shared__ float red[4][256];      // 4KB partial-sum exchange

  const int tid  = blockIdx.x * NTHR + threadIdx.x;
  const int lane = threadIdx.x & 31;
  const int gw   = tid >> 5;
  const int wg   = blockIdx.x;
  const int wv   = threadIdx.x >> 5;          // wave in WG, 0..7
  const int tt   = wv >> 1;                   // local tile 0..3
  const int s    = wv & 1;                    // K-half select
  unsigned bidx  = 0;

  for (int t = 0; t < T_; ++t) {
    const bf16* inp = xT + (size_t)t * (B_ * IN_);

    // ---- phase 1: attn logits = [inp | h] @ attn_W^T + attn_b
    {
      int m0 = (wg >> 3) << 4;
      stage_strip_async(smA0, inp + m0 * IN_);
      stage_strip_async(smA1, hbf + m0 * HID_);
      wait_async_lds();
      __syncthreads();
      int n0 = (((wg & 7) << 2) + tt) << 4;
      f32x8 acc;
      if (s) {
        acc = (f32x8){0.f,0.f,0.f,0.f,0.f,0.f,0.f,0.f};
      } else {
        float bv = attnB[n0 + (lane & 15)];
        acc = (f32x8){bv,bv,bv,bv,bv,bv,bv,bv};
      }
      const bf16* S  = s ? smA1 : smA0;
      const bf16* Wp = s ? attnW + 512 : attnW;
      acc = wmma_acc_lds(acc, S, Wp, 1024, n0, 512, lane);
      if (s) {
        #pragma unroll
        for (int r = 0; r < 8; ++r) red[tt][r * 32 + lane] = acc[r];
      }
      __syncthreads();
      if (!s) {
        #pragma unroll
        for (int r = 0; r < 8; ++r) acc[r] += red[tt][r * 32 + lane];
        store_tile_f32(attn, IN_, m0, n0, acc, lane);
      }
    }
    grid_sync(bar, bidx++);

    // ---- phase 2: row softmax, one wave per batch row
    if (gw < B_) {
      float v[16];
      float mx = -3.402823466e38f;
      #pragma unroll
      for (int i = 0; i < 16; ++i) { v[i] = attn[gw * 512 + lane + 32 * i]; mx = fmaxf(mx, v[i]); }
      #pragma unroll
      for (int sh = 16; sh >= 1; sh >>= 1) mx = fmaxf(mx, __shfl_xor(mx, sh, 32));
      float sum = 0.0f;
      #pragma unroll
      for (int i = 0; i < 16; ++i) { v[i] = __expf(v[i] - mx); sum += v[i]; }
      #pragma unroll
      for (int sh = 16; sh >= 1; sh >>= 1) sum += __shfl_xor(sum, sh, 32);
      float inv = 1.0f / sum;
      #pragma unroll
      for (int i = 0; i < 16; ++i) attn[gw * 512 + lane + 32 * i] = v[i] * inv;
    }
    grid_sync(bar, bidx++);

    // ---- phase 3: applied[b,t'] = sum_m attn[b,m] * x[b,m,t']
    {
      int b  = gw >> 2;
      int t0 = (gw & 3) * 128 + lane * 4;
      const bf16*  xb   = xbf + (size_t)b * IN_ * T_ + t0;
      const float* wrow = attn + b * IN_;
      float a0 = 0.f, a1 = 0.f, a2 = 0.f, a3 = 0.f;
      #pragma unroll 4
      for (int m = 0; m < IN_; ++m) {
        float w = wrow[m];
        bf16x4 xv = *(const bf16x4*)(xb + (size_t)m * T_);
        a0 += w * (float)xv[0]; a1 += w * (float)xv[1];
        a2 += w * (float)xv[2]; a3 += w * (float)xv[3];
      }
      bf16x4 r = { (bf16)a0, (bf16)a1, (bf16)a2, (bf16)a3 };
      *(bf16x4*)(appliedbf + (size_t)b * T_ + t0) = r;
    }
    grid_sync(bar, bidx++);

    // ---- phase 4: g = relu([inp | applied] @ comb_W^T + comb_b) -> bf16
    {
      int m0 = (wg >> 3) << 4;
      stage_strip_async(smA0, inp + m0 * IN_);
      stage_strip_async(smA1, appliedbf + m0 * T_);
      wait_async_lds();
      __syncthreads();
      int n0 = (((wg & 7) << 2) + tt) << 4;
      f32x8 acc;
      if (s) {
        acc = (f32x8){0.f,0.f,0.f,0.f,0.f,0.f,0.f,0.f};
      } else {
        float bv = combB[n0 + (lane & 15)];
        acc = (f32x8){bv,bv,bv,bv,bv,bv,bv,bv};
      }
      const bf16* S  = s ? smA1 : smA0;
      const bf16* Wp = s ? combW + 512 : combW;
      acc = wmma_acc_lds(acc, S, Wp, 1024, n0, 512, lane);
      if (s) {
        #pragma unroll
        for (int r = 0; r < 8; ++r) red[tt][r * 32 + lane] = acc[r];
      }
      __syncthreads();
      if (!s) {
        int n  = n0 + (lane & 15);
        int mb = m0 + ((lane & 16) ? 8 : 0);
        #pragma unroll
        for (int r = 0; r < 8; ++r)
          gbf[(size_t)(mb + r) * HID_ + n] =
              (bf16)fmaxf(acc[r] + red[tt][r * 32 + lane], 0.0f);
      }
    }
    grid_sync(bar, bidx++);

    // ---- phase 5: gi = g @ w_ih^T + b_ih ; gh = h @ w_hh^T + b_hh
    //      16 (matrix,row) strips x 4 WGs each; each wave: 3 tiles, full K.
    {
      int rr  = wg >> 2;               // 0..15
      int mat = rr >> 3;               // 0: gi, 1: gh
      int m0  = (rr & 7) << 4;
      const bf16* A = mat ? hbf : gbf;
      stage_strip_async(smA0, A + m0 * HID_);
      wait_async_lds();
      __syncthreads();
      const bf16*  W    = mat ? whh : wih;
      const float* bias = mat ? bhh : bih;
      float*       C    = mat ? ghm : gi;
      #pragma unroll
      for (int i = 0; i < 3; ++i) {
        int n0 = (((wg & 3) * 24) + wv * 3 + i) << 4;
        float bv = bias[n0 + (lane & 15)];
        f32x8 acc = (f32x8){bv,bv,bv,bv,bv,bv,bv,bv};
        acc = wmma_acc_lds(acc, smA0, W, HID_, n0, HID_, lane);
        store_tile_f32(C, G3_, m0, n0, acc, lane);
      }
    }
    grid_sync(bar, bidx++);

    // ---- phase 6: GRU gate math + state update
    for (int e = tid; e < B_ * HID_; e += NWG * NTHR) {
      int b = e >> 9, j = e & 511;
      const float* gib = gi  + (size_t)b * G3_;
      const float* ghb = ghm + (size_t)b * G3_;
      float r  = 1.0f / (1.0f + __expf(-(gib[j] + ghb[j])));
      float z  = 1.0f / (1.0f + __expf(-(gib[512 + j] + ghb[512 + j])));
      float nn = tanhf(gib[1024 + j] + r * ghb[1024 + j]);
      float ho = h[e];
      float hn = (1.0f - z) * nn + z * ho;
      h[e]   = hn;
      hbf[e] = (bf16)hn;
    }
    grid_sync(bar, bidx++);
  }

  // ---- final: out = h @ out_W^T + out_b  (K=512 split into 256/256)
  {
    int m0 = (wg >> 3) << 4;
    stage_strip_async(smA0, hbf + m0 * HID_);
    wait_async_lds();
    __syncthreads();
    int n0 = (((wg & 7) << 2) + tt) << 4;
    f32x8 acc;
    if (s) {
      acc = (f32x8){0.f,0.f,0.f,0.f,0.f,0.f,0.f,0.f};
    } else {
      float bv = outB[n0 + (lane & 15)];
      acc = (f32x8){bv,bv,bv,bv,bv,bv,bv,bv};
    }
    acc = wmma_acc_lds(acc, smA0 + s * 256, outW + s * 256, OUT_, n0, 256, lane);
    if (s) {
      #pragma unroll
      for (int r = 0; r < 8; ++r) red[tt][r * 32 + lane] = acc[r];
    }
    __syncthreads();
    if (!s) {
      #pragma unroll
      for (int r = 0; r < 8; ++r) acc[r] += red[tt][r * 32 + lane];
      store_tile_f32(out, OUT_, m0, n0, acc, lane);
    }
  }
}

// ---------------------------------------------------------------------------
// Host launcher
// ---------------------------------------------------------------------------
extern "C" void kernel_launch(void* const* d_in, const int* in_sizes, int n_in,
                              void* d_out, int out_size, void* d_ws, size_t ws_size,
                              hipStream_t stream) {
  (void)in_sizes; (void)n_in; (void)out_size; (void)ws_size;
  const float* x      = (const float*)d_in[0];
  const float* attn_W = (const float*)d_in[1];
  const float* attn_b = (const float*)d_in[2];
  const float* comb_W = (const float*)d_in[3];
  const float* comb_b = (const float*)d_in[4];
  const float* w_ih   = (const float*)d_in[5];
  const float* w_hh   = (const float*)d_in[6];
  const float* b_ih   = (const float*)d_in[7];
  const float* b_hh   = (const float*)d_in[8];
  const float* out_W  = (const float*)d_in[9];
  const float* out_b  = (const float*)d_in[10];
  float* out = (float*)d_out;

  char* ws = (char*)d_ws;
  size_t off = 0;
  auto alloc = [&](size_t bytes) -> void* {
    void* p = ws + off;
    off = (off + bytes + 255) & ~(size_t)255;
    return p;
  };

  bf16* xbf       = (bf16*)alloc((size_t)B_ * IN_ * T_ * 2);
  bf16* xT        = (bf16*)alloc((size_t)T_ * B_ * IN_ * 2);
  bf16* attnWb    = (bf16*)alloc((size_t)512 * 1024 * 2);
  bf16* combWb    = (bf16*)alloc((size_t)512 * 1024 * 2);
  bf16* wihb      = (bf16*)alloc((size_t)G3_ * 512 * 2);
  bf16* whhb      = (bf16*)alloc((size_t)G3_ * 512 * 2);
  bf16* outWb     = (bf16*)alloc((size_t)512 * 512 * 2);
  float* hbuf     = (float*)alloc((size_t)B_ * HID_ * 4);
  bf16* hbf       = (bf16*)alloc((size_t)B_ * HID_ * 2);
  float* attnbuf  = (float*)alloc((size_t)B_ * IN_ * 4);
  bf16* appliedbf = (bf16*)alloc((size_t)B_ * T_ * 2);
  bf16* gbf       = (bf16*)alloc((size_t)B_ * HID_ * 2);
  float* gibuf    = (float*)alloc((size_t)B_ * G3_ * 4);
  float* ghbuf    = (float*)alloc((size_t)B_ * G3_ * 4);
  unsigned* bar   = (unsigned*)alloc(64);

  cvt_bf16<<<(512 * 1024 + 255) / 256, 256, 0, stream>>>(attn_W, attnWb, 512 * 1024);
  cvt_bf16<<<(512 * 1024 + 255) / 256, 256, 0, stream>>>(comb_W, combWb, 512 * 1024);
  cvt_bf16<<<(G3_ * 512 + 255) / 256, 256, 0, stream>>>(w_ih, wihb, G3_ * 512);
  cvt_bf16<<<(G3_ * 512 + 255) / 256, 256, 0, stream>>>(w_hh, whhb, G3_ * 512);
  cvt_bf16<<<(512 * 512 + 255) / 256, 256, 0, stream>>>(out_W, outWb, 512 * 512);
  xprep<<<dim3(B_, IN_ / 32, T_ / 32), 256, 0, stream>>>(x, xbf, xT);
  init_state<<<(B_ * HID_ + 255) / 256, 256, 0, stream>>>(hbuf, hbf, bar, B_ * HID_);

  rnn_persistent<<<NWG, NTHR, 0, stream>>>(
      xbf, xT, attnWb, attn_b, combWb, comb_b,
      wihb, b_ih, whhb, b_hh, outWb, out_b,
      hbuf, hbf, attnbuf, appliedbf, gbf, gibuf, ghbuf, out, bar);
}
